// SSIMLoss_25589415149785
// MI455X (gfx1250) — matrix operand order repaired and checked by reading, not compiled, not run on hardware
//
#include <hip/hip_runtime.h>
#include <hip/hip_bf16.h>

// ---------------------------------------------------------------------------
// SSIM loss on (16,3,512,512) f32. Separable 11-tap Gaussian blurs of
// {x, y, x^2, y^2, xy} done with v_wmma_f32_16x16x32_bf16 (banded weight
// matrices), f32 accumulation, fused SSIM map + deterministic reduction.
// LDS layouts chosen so all WMMA operand traffic is b128 DS ops.
// ---------------------------------------------------------------------------

typedef __attribute__((ext_vector_type(16))) __bf16 v16bf;
typedef __attribute__((ext_vector_type(8)))  __bf16 v8bf;
typedef __attribute__((ext_vector_type(8)))  float  v8f;

#define CAT16(lo, hi) __builtin_shufflevector(lo, hi, 0, 1, 2, 3, 4, 5, 6, 7, \
                                              8, 9, 10, 11, 12, 13, 14, 15)

#define IMG      512
#define TW       64          // output tile width
#define THT      32          // output tile height
#define IN_ROWS  48          // 42 valid + zero pad
#define IN_COLS  88          // 74 valid + zero pad; 176B pitch: 16B-aligned,
                             // 44-dword stride -> conflict-free b128 loads
#define HB_ROWSP 56          // transposed hblur row pitch (112B = 7*16B)
#define HB_COLS  64
#define NPLANES  48          // 16 batch * 3 channels
#define NTILES   128         // (512/64) * (512/32)
#define NBLOCKS  (NPLANES * NTILES)       // 6144
#define INV_NPIX (1.0f / 12582912.0f)     // 1/(16*3*512*512)

// Gaussian window, size 11, sigma 1.5, normalized.
__constant__ float c_G[11] = {
    0.00102839f, 0.00759876f, 0.03600376f, 0.10936058f, 0.21300536f,
    0.26601594f,
    0.21300536f, 0.10936058f, 0.03600376f, 0.00759876f, 0.00102839f};

__global__ __launch_bounds__(256)
void SSIMLoss_25589415149785_tile(const float* __restrict__ pred,
                                  const float* __restrict__ targ,
                                  float* __restrict__ partials) {
    __shared__ __align__(16) __bf16 s_x[IN_ROWS][IN_COLS];
    __shared__ __align__(16) __bf16 s_y[IN_ROWS][IN_COLS];
    // horizontally blurred maps, TRANSPOSED: [map][col][row]
    __shared__ __align__(16) __bf16 s_hT[5][HB_COLS][HB_ROWSP];
    __shared__ float s_part[8];

    const int tid  = threadIdx.x;
    const int wave = tid >> 5;
    const int lane = tid & 31;

    const int bid   = blockIdx.x;
    const int plane = bid >> 7;     // / NTILES
    const int t     = bid & 127;
    const int tx    = t & 7;        // tile col (8 across)
    const int ty    = t >> 3;       // tile row (16 down)
    const int C0    = tx * TW;
    const int R0g   = ty * THT;
    const size_t pbase = (size_t)plane * (IMG * IMG);

    // ---- Phase 0: load x, y (5-px halo, zero-padded) into LDS as bf16
    for (int idx = tid; idx < IN_ROWS * IN_COLS; idx += 256) {
        int r = idx / IN_COLS;
        int c = idx - r * IN_COLS;
        int gr = R0g - 5 + r;
        int gc = C0  - 5 + c;
        float xv = 0.0f, yv = 0.0f;
        if (r < 42 && c < 74 && gr >= 0 && gr < IMG && gc >= 0 && gc < IMG) {
            size_t off = pbase + (size_t)gr * IMG + gc;
            xv = pred[off];
            yv = targ[off];
        }
        s_x[r][c] = (__bf16)xv;
        s_y[r][c] = (__bf16)yv;
    }

    // ---- Constant banded weight operands (wB in ISA 7.12.2 k order; wA in contiguous k = 16*(lane>>4) + h)
    // wA (vertical-pass A): A[m][j] = w[j-m], m = lane&15
    // wB (horizontal-pass B): B[j][n] = w[j-n], n = lane&15
    v16bf wA, wB;
    {
        const int mrow = lane & 15;
        const int kAhi = (lane & 16) ? 8  : 0;
        const int kBhi = (lane & 16) ? 16 : 0;
#pragma unroll
        for (int h = 0; h < 16; ++h) {
            int kA = kBhi + h;
            int dA = kA - mrow;
            wA[h] = (__bf16)((dA >= 0 && dA < 11) ? c_G[dA] : 0.0f);
            int kB = ((h & 8) ? 16 : 0) + kAhi + (h & 7);
            int dB = kB - mrow;
            wB[h] = (__bf16)((dB >= 0 && dB < 11) ? c_G[dB] : 0.0f);
        }
    }
    __syncthreads();

    // ---- Phase 1: horizontal blur via WMMA.
    // hb[r][cc] = sum_k w[k] * in[r][cc+k]   (lds col 0 == image col C0-5)
    // A[m][j] = in[rb*16+m][cb*16+j], B = banded weights.
    for (int p = wave; p < 12; p += 8) {           // 3 row-blocks x 4 col-blocks
        const int rb = p >> 2;
        const int cb = p & 3;
        const int rbase = rb * 16 + (lane & 15);
        const int c0 = cb * 16;
        const int klo = (lane & 16) ? 8 : 0;       // A-layout K split per lane

        // A operands: two b128 loads per map, K runs {klo..klo+7, klo+16..+23}
        v8bf xlo = *(const v8bf*)&s_x[rbase][c0 + klo];
        v8bf xhi = *(const v8bf*)&s_x[rbase][c0 + klo + 16];
        v8bf ylo = *(const v8bf*)&s_y[rbase][c0 + klo];
        v8bf yhi = *(const v8bf*)&s_y[rbase][c0 + klo + 16];
        v16bf ax = CAT16(xlo, xhi);
        v16bf ay = CAT16(ylo, yhi);
        v16bf axx, ayy, axy;
#pragma unroll
        for (int h = 0; h < 16; ++h) {             // -> v_pk_mul_bf16
            axx[h] = ax[h] * ax[h];
            ayy[h] = ay[h] * ay[h];
            axy[h] = ax[h] * ay[h];
        }
        const v8f zf = {0.f, 0.f, 0.f, 0.f, 0.f, 0.f, 0.f, 0.f};
        v8f dx  = __builtin_amdgcn_wmma_f32_16x16x32_bf16(false, ax,  false, wB, (short)0, zf, false, false);
        v8f dy  = __builtin_amdgcn_wmma_f32_16x16x32_bf16(false, ay,  false, wB, (short)0, zf, false, false);
        v8f dxx = __builtin_amdgcn_wmma_f32_16x16x32_bf16(false, axx, false, wB, (short)0, zf, false, false);
        v8f dyy = __builtin_amdgcn_wmma_f32_16x16x32_bf16(false, ayy, false, wB, (short)0, zf, false, false);
        v8f dxy = __builtin_amdgcn_wmma_f32_16x16x32_bf16(false, axy, false, wB, (short)0, zf, false, false);

        // D layout: lane holds (m = rb*16 + 8*(lane>>4) + r, n = lane&15).
        // Transposed store: 8 consecutive rows at one col = one b128 store.
        const int n  = lane & 15;
        const int mb = rb * 16 + ((lane >> 4) << 3);
        v8bf p0, p1, p2, p3, p4;
#pragma unroll
        for (int r = 0; r < 8; ++r) {
            p0[r] = (__bf16)dx[r];
            p1[r] = (__bf16)dy[r];
            p2[r] = (__bf16)dxx[r];
            p3[r] = (__bf16)dyy[r];
            p4[r] = (__bf16)dxy[r];
        }
        *(v8bf*)&s_hT[0][c0 + n][mb] = p0;
        *(v8bf*)&s_hT[1][c0 + n][mb] = p1;
        *(v8bf*)&s_hT[2][c0 + n][mb] = p2;
        *(v8bf*)&s_hT[3][c0 + n][mb] = p3;
        *(v8bf*)&s_hT[4][c0 + n][mb] = p4;
    }
    __syncthreads();

    // ---- Phase 2: vertical blur via WMMA + fused SSIM map + reduction.
    float lsum = 0.0f;
    {
        const int rb = wave >> 2;   // 2 row blocks
        const int cb = wave & 3;    // 4 col blocks
        const int r0 = rb * 16;
        const int c0 = cb * 16;
        const int n    = lane & 15;
        const int koff = (lane & 16) ? 16 : 0;
        v8f acc[5];
        const v8f zf = {0.f, 0.f, 0.f, 0.f, 0.f, 0.f, 0.f, 0.f};
#pragma unroll
        for (int m = 0; m < 5; ++m) {
            // B operand: 16 consecutive rows at col c0+n = two b128 loads.
            v8bf blo = *(const v8bf*)&s_hT[m][c0 + n][r0 + koff];
            v8bf bhi = *(const v8bf*)&s_hT[m][c0 + n][r0 + koff + 8];
            v16bf b = CAT16(blo, bhi);
            acc[m] = __builtin_amdgcn_wmma_f32_16x16x32_bf16(false, wA, false, b, (short)0, zf, false, false);
        }
        const float C1 = 0.0004f;   // (0.01*2)^2
        const float C2 = 0.0036f;   // (0.03*2)^2
#pragma unroll
        for (int r = 0; r < 8; ++r) {
            float mux = acc[0][r], muy = acc[1][r];
            float sxx = acc[2][r], syy = acc[3][r], sxy = acc[4][r];
            float mux2 = mux * mux, muy2 = muy * muy, muxy = mux * muy;
            float vx  = fmaxf(sxx - mux2, 0.0f);
            float vy  = fmaxf(syy - muy2, 0.0f);
            float cxy = sxy - muxy;
            float num = (2.0f * muxy + C1) * (2.0f * cxy + C2);
            float den = (mux2 + muy2 + C1) * (vx + vy + C2);
            lsum += num / den;
        }
    }
    // deterministic wave reduce (fixed butterfly order)
#pragma unroll
    for (int off = 16; off > 0; off >>= 1)
        lsum += __shfl_xor(lsum, off, 32);
    if (lane == 0) s_part[wave] = lsum;
    __syncthreads();
    if (tid == 0) {
        float s = 0.0f;
#pragma unroll
        for (int w = 0; w < 8; ++w) s += s_part[w];
        partials[bid] = s;
    }
}

__global__ __launch_bounds__(256)
void SSIMLoss_25589415149785_reduce(const float* __restrict__ partials,
                                    float* __restrict__ out) {
    __shared__ float sh[256];
    const int tid = threadIdx.x;
    float s = 0.0f;
    for (int i = tid; i < NBLOCKS; i += 256) s += partials[i];  // fixed order
    sh[tid] = s;
    __syncthreads();
    for (int off = 128; off > 0; off >>= 1) {
        if (tid < off) sh[tid] += sh[tid + off];
        __syncthreads();
    }
    if (tid == 0) out[0] = 1.0f - sh[0] * INV_NPIX;
}

extern "C" void kernel_launch(void* const* d_in, const int* in_sizes, int n_in,
                              void* d_out, int out_size, void* d_ws, size_t ws_size,
                              hipStream_t stream) {
    const float* pred = (const float*)d_in[0];
    const float* targ = (const float*)d_in[1];
    float* out      = (float*)d_out;
    float* partials = (float*)d_ws;   // NBLOCKS floats = 24 KB scratch

    SSIMLoss_25589415149785_tile<<<NBLOCKS, 256, 0, stream>>>(pred, targ, partials);
    SSIMLoss_25589415149785_reduce<<<1, 256, 0, stream>>>(partials, out);
}
